// NetFixed_34634616275156
// MI455X (gfx1250) — matrix operand order repaired and checked by reading, not compiled, not run on hardware
//
#include <hip/hip_runtime.h>
#include <hip/hip_bf16.h>
#include <stdint.h>

// ---------------------------------------------------------------------------
// math_model constants
// ---------------------------------------------------------------------------
#define BATCH 4096
#define DX    128
#define DD    128
#define NSTEP 64
#define NDLY  16
#define TLEN  81          // N + ND + 1
#define DH    1024
#define KIN   257         // 2*DX + 1
#define KINP  288         // padded to multiple of 32
#define DTIME 0.02f
#define EPSBN 1e-5f

#define CHROWS 64         // rows per bn-partial block
#define NCHUNK (BATCH / CHROWS)   // 64

#if defined(__has_builtin)
#if __has_builtin(__builtin_amdgcn_global_load_async_to_lds_b128)
#define HAVE_ASYNC_LDS 1
#endif
#endif

typedef __attribute__((ext_vector_type(16))) __bf16 v16bf;
typedef __attribute__((ext_vector_type(8)))  float  v8f;
typedef __attribute__((ext_vector_type(4)))  int    v4i;

__device__ __forceinline__ unsigned short f2bf(float f) {
    unsigned u = __float_as_uint(f);
    unsigned r = u + 0x7FFFu + ((u >> 16) & 1u);   // round-to-nearest-even
    return (unsigned short)(r >> 16);
}
__device__ __forceinline__ unsigned pack2bf(float a, float b) {
    return (unsigned)f2bf(a) | ((unsigned)f2bf(b) << 16);
}

union FragBF { v16bf v; uint4 q[2]; };

__device__ __forceinline__ void wait_asynccnt0() {
#if defined(__has_builtin)
#if __has_builtin(__builtin_amdgcn_s_wait_asynccnt)
    __builtin_amdgcn_s_wait_asynccnt(0);
    return;
#else
    asm volatile("s_wait_asynccnt 0" ::: "memory");
    return;
#endif
#else
    asm volatile("s_wait_asynccnt 0" ::: "memory");
#endif
}

// ---------------------------------------------------------------------------
// init d_out: [0,4096) = y_t (=iy0); y region: slots 0..ND = iy0 else 0; z = 0
// ---------------------------------------------------------------------------
__global__ void init_out_kernel(float* __restrict__ out, const float* __restrict__ iy,
                                size_t total) {
    size_t idx = (size_t)blockIdx.x * 256 + threadIdx.x;
    if (idx >= total) return;
    float iy0 = iy[0];
    const size_t yt_end = BATCH;
    const size_t y_end  = BATCH + (size_t)BATCH * TLEN;
    float v = 0.f;
    if (idx < yt_end) v = iy0;
    else if (idx < y_end) {
        size_t r = idx - yt_end;
        int t = (int)(r % TLEN);
        v = (t <= NDLY) ? iy0 : 0.f;
    }
    out[idx] = v;
}

// ---------------------------------------------------------------------------
// weight convert: W [K][N] f32 -> Wt [N][Kpad] bf16 (zero-padded in K)
// ---------------------------------------------------------------------------
__global__ void conv_w_kernel(const float* __restrict__ W, unsigned short* __restrict__ Wt,
                              int K, int N, int Kpad) {
    int idx = blockIdx.x * 256 + threadIdx.x;
    if (idx >= N * Kpad) return;
    int n = idx / Kpad, k = idx % Kpad;
    float v = (k < K) ? W[(size_t)k * N + n] : 0.f;
    Wt[idx] = f2bf(v);
}

// ---------------------------------------------------------------------------
// build inpt f32 [4096][KINP] = concat(x[:,:,i+ND], x[:,:,i], t, zero-pad)
// ---------------------------------------------------------------------------
__global__ void build_input_kernel(const float* __restrict__ x, float* __restrict__ inpt,
                                   int i, float t) {
    int idx = blockIdx.x * 256 + threadIdx.x;
    if (idx >= BATCH * KINP) return;
    int b = idx / KINP, j = idx % KINP;
    float v = 0.f;
    if (j < DX)            v = x[(size_t)b * DX * TLEN + (size_t)j * TLEN + (i + NDLY)];
    else if (j < 2 * DX)   v = x[(size_t)b * DX * TLEN + (size_t)(j - DX) * TLEN + i];
    else if (j == 2 * DX)  v = t;
    inpt[idx] = v;
}

// ---------------------------------------------------------------------------
// BatchNorm stats, phase 1: coalesced partial sums (block owns CHROWS rows,
// thread t owns columns t, t+256, t+512, t+768).
// ---------------------------------------------------------------------------
__global__ __launch_bounds__(256) void bn_partial_kernel(
    const float* __restrict__ A, int K, int Kpad,
    float* __restrict__ psum, float* __restrict__ psq) {
    int chunk = blockIdx.x;
    int tid = threadIdx.x;
    int r0 = chunk * CHROWS;
    float s[4] = {0.f, 0.f, 0.f, 0.f};
    float q[4] = {0.f, 0.f, 0.f, 0.f};
    for (int r = r0; r < r0 + CHROWS; ++r) {
        const float* row = A + (size_t)r * Kpad;
        #pragma unroll
        for (int c = 0; c < 4; ++c) {
            int col = tid + c * 256;
            if (col < K) {
                float v = row[col];
                s[c] += v; q[c] += v * v;
            }
        }
    }
    #pragma unroll
    for (int c = 0; c < 4; ++c) {
        int col = tid + c * 256;
        if (col < K) {
            psum[(size_t)chunk * 1024 + col] = s[c];
            psq [(size_t)chunk * 1024 + col] = q[c];
        }
    }
}

// ---------------------------------------------------------------------------
// BatchNorm stats, phase 2: reduce partials, fold into affine scale/shift.
// ---------------------------------------------------------------------------
__global__ __launch_bounds__(256) void bn_finalize_kernel(
    const float* __restrict__ psum, const float* __restrict__ psq, int K,
    const float* __restrict__ g, const float* __restrict__ b,
    float* __restrict__ scale, float* __restrict__ shift) {
    int col = blockIdx.x * 256 + threadIdx.x;
    if (col >= K) return;
    float s = 0.f, q = 0.f;
    for (int c = 0; c < NCHUNK; ++c) {
        s += psum[(size_t)c * 1024 + col];
        q += psq [(size_t)c * 1024 + col];
    }
    float m   = s * (1.f / BATCH);
    float var = q * (1.f / BATCH) - m * m;
    float rs  = rsqrtf(var + EPSBN);
    float sc  = g[col] * rs;
    scale[col] = sc;
    shift[col] = b[col] - m * sc;
}

// ---------------------------------------------------------------------------
// Apply folded BN affine + f32 -> bf16 conversion: Abf[r][k] (row stride Kpad).
// ---------------------------------------------------------------------------
__global__ __launch_bounds__(256) void bn_norm_kernel(
    const float* __restrict__ A, const float* __restrict__ scale,
    const float* __restrict__ shift, unsigned short* __restrict__ Abf,
    int K, int Kpad) {
    int cid = blockIdx.x * 256 + threadIdx.x;
    int row = blockIdx.y;
    if (cid >= (Kpad >> 2)) return;
    int k = cid << 2;
    float4 v = *(const float4*)(A + (size_t)row * Kpad + k);
    float o[4];
    #pragma unroll
    for (int j = 0; j < 4; ++j) {
        int kk = k + j;
        float sc = 0.f, sh = 0.f;
        if (kk < K) { sc = scale[kk]; sh = shift[kk]; }
        o[j] = ((const float*)&v)[j] * sc + sh;
    }
    uint2 pk;
    pk.x = pack2bf(o[0], o[1]);
    pk.y = pack2bf(o[2], o[3]);
    *(uint2*)(Abf + (size_t)row * Kpad + k) = pk;
}

// ---------------------------------------------------------------------------
// WMMA GEMM: C[M][N] = act( Abf[M][Kpad] @ W + bias )
//   Abf: bf16 normalized activations (row stride Kpad); Wt: bf16 [N][Kpad].
// Block tile 128x128, K-step 32, 8 waves (4 along M x 2 along N),
// double-buffered LDS, one barrier per K-step.
// Staging uses GLOBAL_LOAD_ASYNC_TO_LDS_B128 (no data VGPRs, ASYNCcnt-
// tracked, copy overlapped with WMMA). Per-iteration ordering:
//   wait_asynccnt 0  -> copies into `cur` (issued last iter) complete
//   barrier          -> cur visible to all; all reads of `nxt` done
//   issue async nxt  -> safely after all reads of nxt
//   compute cur      -> overlaps hardware copy of nxt
// ---------------------------------------------------------------------------
#define LDSP 40   // LDS row pitch in bf16 elements (bank-conflict padding)

__global__ __launch_bounds__(256)
__attribute__((amdgpu_waves_per_eu(1)))
void gemm_bf16_kernel(
    const unsigned short* __restrict__ Abf, const unsigned short* __restrict__ Wt,
    const float* __restrict__ bias, float* __restrict__ C,
    int N, int Kpad, int act) {

    __shared__ __align__(16) unsigned short Asm[2][128 * LDSP];
    __shared__ __align__(16) unsigned short Bsm[2][128 * LDSP];

    const int tid  = threadIdx.x;
    const int lane = tid & 31;
    const int wid  = tid >> 5;
    const int wm   = wid & 3;          // 4 waves along M
    const int wn   = wid >> 2;         // 2 waves along N
    const int half = lane >> 4;
    const int l16  = lane & 15;
    const int bm   = blockIdx.x * 128;
    const int bn   = blockIdx.y * 128;

    // staging coordinates: 512 uint4 chunks per tile, 2 per thread
    const int srow = tid >> 2;              // rows {srow, srow+64}
    const int skk  = (tid & 3) << 3;        // bf16 offset within 32-k row

    v8f acc[2][4] = {};
    const int ksteps = Kpad >> 5;

#ifdef HAVE_ASYNC_LDS
    typedef __attribute__((address_space(1))) v4i gv4i;
    typedef __attribute__((address_space(3))) v4i lv4i;
    auto asyncTile = [&](int buf, int kbase) {
        #pragma unroll
        for (int it = 0; it < 2; ++it) {
            int r = srow + it * 64;
            __builtin_amdgcn_global_load_async_to_lds_b128(
                (gv4i*)(Abf + (size_t)(bm + r) * Kpad + kbase + skk),
                (lv4i*)&Asm[buf][r * LDSP + skk], 0, 0);
            __builtin_amdgcn_global_load_async_to_lds_b128(
                (gv4i*)(Wt + (size_t)(bn + r) * Kpad + kbase + skk),
                (lv4i*)&Bsm[buf][r * LDSP + skk], 0, 0);
        }
    };
    asyncTile(0, 0);
#else
    uint4 aReg[2], bReg[2];
    auto loadG = [&](int kbase) {
        #pragma unroll
        for (int it = 0; it < 2; ++it) {
            int r = srow + it * 64;
            aReg[it] = *(const uint4*)(Abf + (size_t)(bm + r) * Kpad + kbase + skk);
            bReg[it] = *(const uint4*)(Wt  + (size_t)(bn + r) * Kpad + kbase + skk);
        }
    };
    auto storeL = [&](int buf) {
        #pragma unroll
        for (int it = 0; it < 2; ++it) {
            int r = srow + it * 64;
            *(uint4*)&Asm[buf][r * LDSP + skk] = aReg[it];
            *(uint4*)&Bsm[buf][r * LDSP + skk] = bReg[it];
        }
    };
    loadG(0);
    storeL(0);
#endif

    for (int ks = 0; ks < ksteps; ++ks) {
        const int cur = ks & 1;
#ifdef HAVE_ASYNC_LDS
        wait_asynccnt0();                 // copies into `cur` complete
        __syncthreads();                  // visible to all; reads of nxt done
        if (ks + 1 < ksteps) asyncTile(cur ^ 1, (ks + 1) << 5);
#else
        if (ks + 1 < ksteps) loadG((ks + 1) << 5);   // prefetch next tile
        __syncthreads();                              // buf `cur` stores visible
#endif

        // ---- fragments (ISA 7.12.2 layouts) ----
        FragBF af[2], bfr[4];
        #pragma unroll
        for (int mi = 0; mi < 2; ++mi) {
            int row = wm * 32 + mi * 16 + l16;
            const unsigned short* p = &Asm[cur][row * LDSP + 8 * half];
            af[mi].q[0] = *(const uint4*)p;            // K = 8*half + 0..7
            af[mi].q[1] = *(const uint4*)(p + 16);     // K = 16 + 8*half + 0..7
        }
        #pragma unroll
        for (int ni = 0; ni < 4; ++ni) {
            int n = wn * 64 + ni * 16 + l16;
            const unsigned short* p = &Bsm[cur][n * LDSP + 8 * half];
            bfr[ni].q[0] = *(const uint4*)p;           // K = 8*half + 0..7
            bfr[ni].q[1] = *(const uint4*)(p + 16);    // K = 16 + 8*half + 0..7
        }

        // ---- 8x WMMA ----
        #pragma unroll
        for (int mi = 0; mi < 2; ++mi)
            #pragma unroll
            for (int ni = 0; ni < 4; ++ni)
                acc[mi][ni] = __builtin_amdgcn_wmma_f32_16x16x32_bf16(
                    false, af[mi].v, false, bfr[ni].v,
                    (short)0, acc[mi][ni], false, false);

#ifndef HAVE_ASYNC_LDS
        if (ks + 1 < ksteps) storeL(cur ^ 1);
#endif
    }

    // ---- epilogue: bias + optional tanh, f32 store ----
    #pragma unroll
    for (int mi = 0; mi < 2; ++mi) {
        #pragma unroll
        for (int ni = 0; ni < 4; ++ni) {
            int col = bn + wn * 64 + ni * 16 + l16;
            float bv = bias[col];
            #pragma unroll
            for (int v = 0; v < 8; ++v) {
                int row = bm + wm * 32 + mi * 16 + half * 8 + v;
                float val = acc[mi][ni][v] + bv;
                if (act) val = tanhf(val);
                C[(size_t)row * N + col] = val;
            }
        }
    }
}

// ---------------------------------------------------------------------------
// driver + Euler update, one block per batch row, 128 threads (one per DD)
// ---------------------------------------------------------------------------
__global__ __launch_bounds__(128) void step_update_kernel(
    const float* __restrict__ x, const float* __restrict__ dW,
    const float* __restrict__ zt, float* __restrict__ yt,
    float* __restrict__ y, float* __restrict__ z, int i) {
    int b = blockIdx.x;
    int d = threadIdx.x;
    int it = i + NDLY;

    float ztv = zt[(size_t)b * DD + d];
    float zdv = z[(size_t)b * DD * TLEN + (size_t)d * TLEN + i];
    float dwv = dW[(size_t)b * DD * NSTEP + (size_t)d * NSTEP + i];
    float xv  = x[(size_t)b * DX * TLEN + (size_t)d * TLEN + it];

    __shared__ float s1[128], s2[128], s3[128];
    s1[d] = ztv * zdv; s2[d] = ztv * dwv; s3[d] = xv;
    __syncthreads();
    for (int o = 64; o > 0; o >>= 1) {
        if (d < o) { s1[d] += s1[d + o]; s2[d] += s2[d + o]; s3[d] += s3[d + o]; }
        __syncthreads();
    }
    z[(size_t)b * DD * TLEN + (size_t)d * TLEN + (it + 1)] = ztv;
    if (d == 0) {
        float y_t = yt[b];
        float y_d = y[(size_t)b * TLEN + i];
        float drv = -y_t + 0.1f * y_d + s1[0] + 0.01f * s3[0];
        float ynew = y_t - drv * DTIME + s2[0];
        yt[b] = ynew;
        y[(size_t)b * TLEN + (it + 1)] = ynew;
    }
}

// ---------------------------------------------------------------------------
// host orchestration
// ---------------------------------------------------------------------------
extern "C" void kernel_launch(void* const* d_in, const int* in_sizes, int n_in,
                              void* d_out, int out_size, void* d_ws, size_t ws_size,
                              hipStream_t stream) {
    const float* x        = (const float*)d_in[0];
    const float* dW_f     = (const float*)d_in[1];
    const float* iy       = (const float*)d_in[2];
    const float* bn_in_g  = (const float*)d_in[3];
    const float* bn_in_b  = (const float*)d_in[4];
    const float* W_in     = (const float*)d_in[5];
    const float* b_in     = (const float*)d_in[6];
    const float* Ws_h     = (const float*)d_in[7];
    const float* bs_h     = (const float*)d_in[8];
    const float* bns_g    = (const float*)d_in[9];
    const float* bns_b    = (const float*)d_in[10];
    const float* bn_out_g = (const float*)d_in[11];
    const float* bn_out_b = (const float*)d_in[12];
    const float* W_out    = (const float*)d_in[13];
    const float* b_out    = (const float*)d_in[14];

    float* out = (float*)d_out;
    float* yt  = out;
    float* yb  = out + BATCH;
    float* zb  = out + BATCH + (size_t)BATCH * TLEN;

    // workspace carve-up (256B aligned)
    char* ws = (char*)d_ws;
    size_t off = 0;
    auto carve = [&](size_t bytes) {
        size_t r = off;
        off = (off + bytes + 255) & ~(size_t)255;
        return r;
    };
    float* inpt = (float*)(ws + carve((size_t)BATCH * KINP * sizeof(float)));
    float* s0   = (float*)(ws + carve((size_t)BATCH * DH * sizeof(float)));
    float* s1   = (float*)(ws + carve((size_t)BATCH * DH * sizeof(float)));
    float* ztb  = (float*)(ws + carve((size_t)BATCH * DD * sizeof(float)));
    float* scl  = (float*)(ws + carve((size_t)DH * sizeof(float)));
    float* shf  = (float*)(ws + carve((size_t)DH * sizeof(float)));
    float* psum = (float*)(ws + carve((size_t)NCHUNK * 1024 * sizeof(float)));
    float* psq  = (float*)(ws + carve((size_t)NCHUNK * 1024 * sizeof(float)));
    unsigned short* Abf   = (unsigned short*)(ws + carve((size_t)BATCH * DH * 2));
    unsigned short* WtIn  = (unsigned short*)(ws + carve((size_t)DH * KINP * 2));
    unsigned short* WtH   = (unsigned short*)(ws + carve((size_t)3 * DH * DH * 2));
    unsigned short* WtOut = (unsigned short*)(ws + carve((size_t)DD * DH * 2));

    // --- init outputs ---
    {
        size_t total = (size_t)BATCH + (size_t)BATCH * TLEN + (size_t)BATCH * DD * TLEN;
        init_out_kernel<<<(unsigned)((total + 255) / 256), 256, 0, stream>>>(out, iy, total);
    }
    // --- weight conversion (bf16, transposed, K-padded) ---
    conv_w_kernel<<<(DH * KINP + 255) / 256, 256, 0, stream>>>(W_in, WtIn, KIN, DH, KINP);
    for (int l = 0; l < 3; ++l)
        conv_w_kernel<<<(DH * DH + 255) / 256, 256, 0, stream>>>(
            Ws_h + (size_t)l * DH * DH, WtH + (size_t)l * DH * DH, DH, DH, DH);
    conv_w_kernel<<<(DD * DH + 255) / 256, 256, 0, stream>>>(W_out, WtOut, DH, DD, DH);

    const dim3 blk(256);
    const dim3 gridH(BATCH / 128, DH / 128);   // (32, 8)
    const dim3 gridO(BATCH / 128, DD / 128);   // (32, 1)

    // stats + normalize: f32 activations -> folded-BN bf16 buffer Abf
    auto bn = [&](const float* Abuf, int K, int Kpad, const float* g, const float* b) {
        bn_partial_kernel<<<NCHUNK, blk, 0, stream>>>(Abuf, K, Kpad, psum, psq);
        bn_finalize_kernel<<<(K + 255) / 256, blk, 0, stream>>>(psum, psq, K, g, b, scl, shf);
        dim3 gnorm(((Kpad >> 2) + 255) / 256, BATCH);
        bn_norm_kernel<<<gnorm, blk, 0, stream>>>(Abuf, scl, shf, Abf, K, Kpad);
    };

    for (int i = 0; i < NSTEP; ++i) {
        float t = (float)i * DTIME;

        build_input_kernel<<<(BATCH * KINP + 255) / 256, blk, 0, stream>>>(x, inpt, i, t);

        // layer 0: inpt -> s0
        bn(inpt, KIN, KINP, bn_in_g, bn_in_b);
        gemm_bf16_kernel<<<gridH, blk, 0, stream>>>(Abf, WtIn, b_in, s0, DH, KINP, 1);
        // hidden layers
        float* cur = s0;
        float* nxt = s1;
        for (int l = 0; l < 3; ++l) {
            bn(cur, DH, DH, bns_g + (size_t)l * DH, bns_b + (size_t)l * DH);
            gemm_bf16_kernel<<<gridH, blk, 0, stream>>>(Abf, WtH + (size_t)l * DH * DH,
                                                        bs_h + (size_t)l * DH, nxt,
                                                        DH, DH, 1);
            float* tmp = cur; cur = nxt; nxt = tmp;
        }
        // output layer (no tanh): cur -> ztb
        bn(cur, DH, DH, bn_out_g, bn_out_b);
        gemm_bf16_kernel<<<gridO, blk, 0, stream>>>(Abf, WtOut, b_out, ztb, DD, DH, 0);
        // driver + Euler update
        step_update_kernel<<<BATCH, 128, 0, stream>>>(x, dW_f, ztb, yt, yb, zb, i);
    }
    (void)in_sizes; (void)n_in; (void)out_size; (void)ws_size;
}
